// DecoderRNN_65455301591781
// MI455X (gfx1250) — matrix operand order repaired and checked by reading, not compiled, not run on hardware
//
#include <hip/hip_runtime.h>
#include <hip/hip_bf16.h>

typedef __bf16 bf16;
typedef __attribute__((ext_vector_type(16))) __bf16 bf16x16;
typedef __attribute__((ext_vector_type(8)))  __bf16 bf16x8;
typedef __attribute__((ext_vector_type(8)))  float  f32x8;

#define B_ 128
#define T_ 31
#define E_ 512
#define H_ 512
#define V_ 10000
#define A_ 2048

// A fragment (16x32 bf16): lane = row M, contiguous 16 K per half-wave:
//   VGPR0-3: K = hl*16 + 0..7 ; VGPR4-7: K = hl*16 + 8..15
__device__ __forceinline__ bf16x16 load_afrag(const bf16* __restrict__ arow, int k, int hl)
{
    bf16x8 lo = *(const bf16x8*)(arow + k + hl * 16);
    bf16x8 hi = *(const bf16x8*)(arow + k + hl * 16 + 8);
    bf16x16 a;
    #pragma unroll
    for (int i = 0; i < 8; ++i) { a[i] = lo[i]; a[8 + i] = hi[i]; }
    return a;
}

// ---------------------------------------------------------------------------
// Generic bf16 WMMA GEMM:  C[M,N] = concat(A0[M,K0], A1[M,K1]) * Bw[N,K]^T + bias
// A rows are K-contiguous (activations); Bw rows are K-contiguous (PyTorch
// weight layout [out,in]), so the dot runs over the contiguous dim for both.
// One wave computes a 32(M) x 64(N) tile: 8 accumulators, each B fragment
// reused 2x, each A fragment reused 4x (8 WMMAs per 6 fragment loads).
// K0, K1 multiples of 32; M multiple of 32; N multiple of 16.
// The concat is handled as two sequential K loops (no per-iteration selects).
// ---------------------------------------------------------------------------
__global__ __launch_bounds__(256) void gemm_bf16_wmma(
    const bf16* __restrict__ A0, int lda0, int K0,
    const bf16* __restrict__ A1, int lda1, int K1,
    const bf16* __restrict__ Bw, int ldb,
    const float* __restrict__ bias,
    float* __restrict__ Cf, bf16* __restrict__ Cbf,
    int M, int N)
{
    const int tilesM  = M >> 5;
    const int tilesN4 = ((N >> 4) + 3) >> 2;
    const int gw = (blockIdx.x * 256 + threadIdx.x) >> 5;   // wave id (uniform per wave)
    if (gw >= tilesM * tilesN4) return;                     // whole-wave exit: EXEC all-1s

    const int tm   = gw % tilesM;
    const int tn4  = gw / tilesM;
    const int lane = threadIdx.x & 31;
    const int hl   = lane >> 4;      // half-wave select
    const int m    = lane & 15;

    const int Mb  = tm << 5;
    const int Nb0 = tn4 << 6;

    // Column base per 16-wide fragment; clamp duplicates when N % 64 != 0.
    const bf16* brow[4];
    #pragma unroll
    for (int f = 0; f < 4; ++f) {
        int nbf = Nb0 + (f << 4);
        int nb  = (nbf + 16 <= N) ? nbf : (N - 16);
        brow[f] = Bw + (size_t)(nb + m) * ldb;
    }

    f32x8 zero = {0.f, 0.f, 0.f, 0.f, 0.f, 0.f, 0.f, 0.f};
    f32x8 acc0[4] = {zero, zero, zero, zero};   // rows Mb .. Mb+15
    f32x8 acc1[4] = {zero, zero, zero, zero};   // rows Mb+16 .. Mb+31

    // ---- segment 0 ----
    {
        const bf16* ar0 = A0 + (size_t)(Mb + m) * lda0;
        const bf16* ar1 = A0 + (size_t)(Mb + 16 + m) * lda0;
        for (int kb = 0; kb < K0; kb += 32) {
            bf16x16 a0 = load_afrag(ar0, kb, hl);
            bf16x16 a1 = load_afrag(ar1, kb, hl);
            #pragma unroll
            for (int f = 0; f < 4; ++f) {
                // B fragment (32x16): lane = column N, contiguous 16 K per half-wave.
                bf16x16 b = *(const bf16x16*)(brow[f] + kb + hl * 16);
                acc0[f] = __builtin_amdgcn_wmma_f32_16x16x32_bf16(
                    false, a0, false, b, (short)0, acc0[f], false, false);
                acc1[f] = __builtin_amdgcn_wmma_f32_16x16x32_bf16(
                    false, a1, false, b, (short)0, acc1[f], false, false);
            }
        }
    }
    // ---- segment 1 (concat) ----
    if (A1) {
        const bf16* ar0 = A1 + (size_t)(Mb + m) * lda1;
        const bf16* ar1 = A1 + (size_t)(Mb + 16 + m) * lda1;
        for (int kb = 0; kb < K1; kb += 32) {
            bf16x16 a0 = load_afrag(ar0, kb, hl);
            bf16x16 a1 = load_afrag(ar1, kb, hl);
            #pragma unroll
            for (int f = 0; f < 4; ++f) {
                bf16x16 b = *(const bf16x16*)(brow[f] + K0 + kb + hl * 16);
                acc0[f] = __builtin_amdgcn_wmma_f32_16x16x32_bf16(
                    false, a0, false, b, (short)0, acc0[f], false, false);
                acc1[f] = __builtin_amdgcn_wmma_f32_16x16x32_bf16(
                    false, a1, false, b, (short)0, acc1[f], false, false);
            }
        }
    }

    // C/D layout: VGPR r -> row (base + r + 8*hl), column = nb + m.
    #pragma unroll
    for (int f = 0; f < 4; ++f) {
        int nbf = Nb0 + (f << 4);
        if (nbf >= N) break;                 // clamped duplicate tile: skip store
        int col = nbf + m;
        float bv = bias ? bias[col] : 0.f;
        #pragma unroll
        for (int r = 0; r < 8; ++r) {
            int row0 = Mb + r + hl * 8;
            int row1 = Mb + 16 + r + hl * 8;
            float v0 = acc0[f][r] + bv;
            float v1 = acc1[f][r] + bv;
            if (Cf) {
                Cf[(size_t)row0 * N + col] = v0;
                Cf[(size_t)row1 * N + col] = v1;
            }
            if (Cbf) {
                Cbf[(size_t)row0 * N + col] = (bf16)v0;
                Cbf[(size_t)row1 * N + col] = (bf16)v1;
            }
        }
    }
}

// ---------------------------------------------------------------------------
// Softmax over A_=2048 attention logits per row, fused with attended = cnn * aw.
// One block (256 threads) per batch row.
// ---------------------------------------------------------------------------
__global__ __launch_bounds__(256) void attn_softmax(
    const float* __restrict__ logits,   // [B, A]
    const float* __restrict__ cnnf,     // [B, A]
    bf16* __restrict__ attended)        // [B, A]
{
    __shared__ float red[256];
    const int b = blockIdx.x;
    const int tid = threadIdx.x;
    const float* row = logits + (size_t)b * A_;

    float mx = -3.4e38f;
    for (int i = tid; i < A_; i += 256) mx = fmaxf(mx, row[i]);
    red[tid] = mx; __syncthreads();
    for (int s = 128; s > 0; s >>= 1) {
        if (tid < s) red[tid] = fmaxf(red[tid], red[tid + s]);
        __syncthreads();
    }
    mx = red[0]; __syncthreads();

    float sum = 0.f;
    for (int i = tid; i < A_; i += 256) sum += __expf(row[i] - mx);
    red[tid] = sum; __syncthreads();
    for (int s = 128; s > 0; s >>= 1) {
        if (tid < s) red[tid] += red[tid + s];
        __syncthreads();
    }
    float inv = 1.f / red[0];

    for (int i = tid; i < A_; i += 256) {
        float aw = __expf(row[i] - mx) * inv;
        attended[(size_t)b * A_ + i] = (bf16)(cnnf[(size_t)b * A_ + i] * aw);
    }
}

// ---------------------------------------------------------------------------
// LSTM cell elementwise: gates [B,4H] (i,f,g,o), fp32 state c, bf16 h out,
// also records h into packed hidden[t*B + b][H] for the final logits GEMM.
// ---------------------------------------------------------------------------
__global__ void lstm_cell(const float* __restrict__ gates,
                          float* __restrict__ c,
                          bf16* __restrict__ h_bf,
                          bf16* __restrict__ hidden,
                          int t)
{
    int idx = blockIdx.x * blockDim.x + threadIdx.x;
    if (idx >= B_ * H_) return;
    int b = idx / H_, hh = idx % H_;
    const float* g = gates + (size_t)b * 4 * H_;
    float gi = 1.f / (1.f + __expf(-g[hh]));
    float gf = 1.f / (1.f + __expf(-g[H_ + hh]));
    float gg = tanhf(g[2 * H_ + hh]);
    float go = 1.f / (1.f + __expf(-g[3 * H_ + hh]));
    float c2 = gf * c[idx] + gi * gg;
    float h2 = go * tanhf(c2);
    c[idx] = c2;
    h_bf[idx] = (bf16)h2;
    hidden[(size_t)t * B_ * H_ + idx] = (bf16)h2;
}

// Build X[t][b][e] bf16: t==0 -> image features, else embedding of captions[b][t-1].
__global__ void build_x(const float* __restrict__ features,
                        const float* __restrict__ embed,
                        const int* __restrict__ captions,
                        bf16* __restrict__ X)
{
    int idx = blockIdx.x * blockDim.x + threadIdx.x;
    if (idx >= (T_ + 1) * B_ * E_) return;
    int t = idx / (B_ * E_);
    int r = idx % (B_ * E_);
    int b = r / E_, e = r % E_;
    float v;
    if (t == 0) v = features[(size_t)b * E_ + e];
    else {
        int tok = captions[(size_t)b * T_ + (t - 1)];
        v = embed[(size_t)tok * E_ + e];
    }
    X[idx] = (bf16)v;
}

__global__ void f32_to_bf16_strided(const float* __restrict__ src, int rows, int cols,
                                    bf16* __restrict__ dst, int ld, int coloff)
{
    size_t idx = (size_t)blockIdx.x * blockDim.x + threadIdx.x;
    if (idx >= (size_t)rows * cols) return;
    int r = (int)(idx / cols), cc = (int)(idx % cols);
    dst[(size_t)r * ld + coloff + cc] = (bf16)src[idx];
}

__global__ void add_vec(const float* __restrict__ a, const float* __restrict__ b,
                        float* __restrict__ o, int n)
{
    int i = blockIdx.x * blockDim.x + threadIdx.x;
    if (i < n) o[i] = a[i] + b[i];
}

__global__ void zero_bf16(bf16* p, int n)
{
    int i = blockIdx.x * blockDim.x + threadIdx.x;
    if (i < n) p[i] = (bf16)0.f;
}

__global__ void zero_f32(float* p, int n)
{
    int i = blockIdx.x * blockDim.x + threadIdx.x;
    if (i < n) p[i] = 0.f;
}

extern "C" void kernel_launch(void* const* d_in, const int* in_sizes, int n_in,
                              void* d_out, int out_size, void* d_ws, size_t ws_size,
                              hipStream_t stream) {
    (void)in_sizes; (void)n_in; (void)out_size; (void)ws_size;

    const float* features = (const float*)d_in[0];
    const float* cnnf     = (const float*)d_in[1];
    const int*   captions = (const int*)d_in[2];
    /* d_in[3] lengths: constant T+1, unused */
    const float* embed    = (const float*)d_in[4];
    const float* W_ih     = (const float*)d_in[5];
    const float* W_hh     = (const float*)d_in[6];
    const float* b_ih     = (const float*)d_in[7];
    const float* b_hh     = (const float*)d_in[8];
    const float* W_attn   = (const float*)d_in[9];
    const float* b_attn   = (const float*)d_in[10];
    const float* W_attd   = (const float*)d_in[11];
    const float* b_attd   = (const float*)d_in[12];
    const float* W_out    = (const float*)d_in[13];
    const float* b_out    = (const float*)d_in[14];
    float* out = (float*)d_out;

    // ---- workspace carve-up (256B aligned) ----
    char* wsb = (char*)d_ws;
    size_t off = 0;
    auto alloc = [&](size_t bytes) {
        void* p = wsb + off;
        off = (off + bytes + 255) & ~(size_t)255;
        return p;
    };
    bf16*  Xbf      = (bf16*) alloc((size_t)(T_ + 1) * B_ * E_ * sizeof(bf16));
    bf16*  Wcat_bf  = (bf16*) alloc((size_t)4 * H_ * (E_ + H_) * sizeof(bf16)); // [4H, E+H]
    bf16*  Wattn_bf = (bf16*) alloc((size_t)A_ * (E_ + H_) * sizeof(bf16));     // [A, E+H]
    bf16*  Wattd_bf = (bf16*) alloc((size_t)E_ * (A_ + E_) * sizeof(bf16));     // [E, E+A]
    bf16*  Wout_bf  = (bf16*) alloc((size_t)V_ * H_ * sizeof(bf16));            // [V, H]
    float* bcat     = (float*)alloc((size_t)4 * H_ * sizeof(float));
    float* gates    = (float*)alloc((size_t)B_ * 4 * H_ * sizeof(float));
    float* attnlog  = (float*)alloc((size_t)B_ * A_ * sizeof(float));
    bf16*  attended = (bf16*) alloc((size_t)B_ * A_ * sizeof(bf16));
    bf16*  x2bf     = (bf16*) alloc((size_t)B_ * E_ * sizeof(bf16));
    bf16*  h_bf     = (bf16*) alloc((size_t)B_ * H_ * sizeof(bf16));
    float* c_st     = (float*)alloc((size_t)B_ * H_ * sizeof(float));
    bf16*  hidden   = (bf16*) alloc((size_t)(T_ + 1) * B_ * H_ * sizeof(bf16));

    auto blks = [](size_t n) { return (unsigned)((n + 255) / 256); };
    auto gemm = [&](const bf16* A0, int lda0, int K0,
                    const bf16* A1, int lda1, int K1,
                    const bf16* Bw, int ldb, const float* bias,
                    float* Cf, bf16* Cbf, int M, int N) {
        int tilesM  = M >> 5;
        int tilesN4 = ((N >> 4) + 3) >> 2;
        int waves   = tilesM * tilesN4;
        int nblocks = (waves + 7) / 8;       // 8 waves (256 threads) per block
        gemm_bf16_wmma<<<nblocks, 256, 0, stream>>>(A0, lda0, K0, A1, lda1, K1,
                                                    Bw, ldb, bias, Cf, Cbf, M, N);
    };

    // ---- one-time (per launch) weight conversion to bf16 ----
    f32_to_bf16_strided<<<blks((size_t)4 * H_ * E_), 256, 0, stream>>>(
        W_ih, 4 * H_, E_, Wcat_bf, E_ + H_, 0);
    f32_to_bf16_strided<<<blks((size_t)4 * H_ * H_), 256, 0, stream>>>(
        W_hh, 4 * H_, H_, Wcat_bf, E_ + H_, E_);
    f32_to_bf16_strided<<<blks((size_t)A_ * (E_ + H_)), 256, 0, stream>>>(
        W_attn, A_, E_ + H_, Wattn_bf, E_ + H_, 0);
    f32_to_bf16_strided<<<blks((size_t)E_ * (A_ + E_)), 256, 0, stream>>>(
        W_attd, E_, A_ + E_, Wattd_bf, A_ + E_, 0);
    f32_to_bf16_strided<<<blks((size_t)V_ * H_), 256, 0, stream>>>(
        W_out, V_, H_, Wout_bf, H_, 0);
    add_vec<<<blks(4 * H_), 256, 0, stream>>>(b_ih, b_hh, bcat, 4 * H_);

    // ---- state init + input assembly ----
    zero_bf16<<<blks(B_ * H_), 256, 0, stream>>>(h_bf, B_ * H_);
    zero_f32 <<<blks(B_ * H_), 256, 0, stream>>>(c_st, B_ * H_);
    build_x<<<blks((size_t)(T_ + 1) * B_ * E_), 256, 0, stream>>>(
        features, embed, captions, Xbf);

    // ---- timestep 0: plain LSTM step (h=0 via zeroed h_bf) ----
    gemm(Xbf, E_, E_, h_bf, H_, H_, Wcat_bf, E_ + H_, bcat, gates, nullptr, B_, 4 * H_);
    lstm_cell<<<blks(B_ * H_), 256, 0, stream>>>(gates, c_st, h_bf, hidden, 0);

    // ---- timesteps 1..T: attention + LSTM ----
    for (int t = 1; t <= T_; ++t) {
        const bf16* xt = Xbf + (size_t)t * B_ * E_;
        // attention logits = [x_t, h] @ W_attn^T + b_attn   (M=128, N=2048, K=1024)
        gemm(xt, E_, E_, h_bf, H_, H_, Wattn_bf, E_ + H_, b_attn, attnlog, nullptr, B_, A_);
        attn_softmax<<<B_, 256, 0, stream>>>(attnlog, cnnf, attended);
        // x2 = [x_t, attended] @ W_attd^T + b_attd          (M=128, N=512, K=2560)
        gemm(xt, E_, E_, attended, A_, A_, Wattd_bf, A_ + E_, b_attd, nullptr, x2bf, B_, E_);
        // gates = [x2, h] @ [W_ih | W_hh]^T + (b_ih+b_hh)   (M=128, N=2048, K=1024)
        gemm(x2bf, E_, E_, h_bf, H_, H_, Wcat_bf, E_ + H_, bcat, gates, nullptr, B_, 4 * H_);
        lstm_cell<<<blks(B_ * H_), 256, 0, stream>>>(gates, c_st, h_bf, hidden, t);
    }

    // ---- logits = hidden @ W_out^T + b_out  (M=4096, N=10000, K=512) ----
    gemm(hidden, H_, H_, nullptr, 0, 0, Wout_bf, H_, b_out, out, nullptr,
         (T_ + 1) * B_, V_);
}
